// MXMGlobalMP_24953759989848
// MI455X (gfx1250) — compile-verified, matrix operand order checked by reading, not run on hardware
//
#include <hip/hip_runtime.h>

// MXMNet global message-passing layer for MI455X (gfx1250, wave32, WMMA).
// GEMMs on V_WMMA_F32_16X16X32_F16 (f16 in, f32 accumulate).
// - Weights staged in LDS pre-swizzled into the WMMA B register layout
//   (2x ds_load_b128 per B tile per lane).
// - Node features converted to f16 once per propagate; edge kernel gathers
//   them directly in A layout with 2x global_load_b128 per tile (halves L2
//   gather traffic and kills most v_cvt VALU work in the hot kernel).
// ws layout (floats): bufA[N*64] bufB[N*64] bufC[N*64] ACC[N*64] CNT[N]
//                     then H16[N*64 halves]  (~58 MB total)

#define DD 64

typedef __attribute__((ext_vector_type(16))) _Float16 v16h;
typedef __attribute__((ext_vector_type(8)))  _Float16 v8h;
typedef __attribute__((ext_vector_type(8)))  float    v8f;

__device__ __forceinline__ v8f wmma16(v16h a, v16h b, v8f c) {
  return __builtin_amdgcn_wmma_f32_16x16x32_f16(false, a, false, b, (short)0, c,
                                                false, false);
}

// Fast silu: x * v_rcp_f32(1 + exp(-x))
__device__ __forceinline__ float silu_f(float x) {
  return x * __builtin_amdgcn_rcpf(1.0f + __expf(-x));
}

__device__ __forceinline__ v16h cvt16(float4 a, float4 b, float4 c, float4 d) {
  v16h v;
  v[0] = (_Float16)a.x;  v[1] = (_Float16)a.y;  v[2] = (_Float16)a.z;  v[3] = (_Float16)a.w;
  v[4] = (_Float16)b.x;  v[5] = (_Float16)b.y;  v[6] = (_Float16)b.z;  v[7] = (_Float16)b.w;
  v[8] = (_Float16)c.x;  v[9] = (_Float16)c.y;  v[10] = (_Float16)c.z; v[11] = (_Float16)c.w;
  v[12] = (_Float16)d.x; v[13] = (_Float16)d.y; v[14] = (_Float16)d.z; v[15] = (_Float16)d.w;
  return v;
}

// 16x32 f16 A-tile from per-lane f32 row pointer (b128 loads + cvt).
// CDNA5 16-bit A layout: lanes 0-15 hold M=lane with K in {0..7,16..23},
// lanes 16-31 hold the same rows with K in {8..15,24..31}.
__device__ __forceinline__ v16h load_a_row_f32(const float* rowp, int k0) {
  const int lane = threadIdx.x & 31;
  const int kb = k0 + ((lane >> 4) << 3);
  const float4* p = (const float4*)(rowp + kb);
  return cvt16(p[0], p[1], p[4], p[5]);   // kb+0..7 and kb+16..23
}

// Same tile from an f16 row pointer: two global_load_b128, zero converts.
__device__ __forceinline__ v16h load_a_row_f16(const _Float16* rowp, int k0) {
  const int lane = threadIdx.x & 31;
  const int kb = k0 + ((lane >> 4) << 3);
  const v8h lo = *(const v8h*)(rowp + kb);
  const v8h hi = *(const v8h*)(rowp + kb + 16);
  v16h a;
#pragma unroll
  for (int i = 0; i < 8; ++i) { a[i] = lo[i]; a[8 + i] = hi[i]; }
  return a;
}

// Stage W (KxN row-major f32, K = ktiles*32, N = 64) into LDS pre-swizzled so
// tile t = kt*4+nt lives at dst[(t*32+lane)*16 .. +15] in WMMA B layout.
__device__ __forceinline__ void stage_b_swizzled(_Float16* dst, const float* W,
                                                 int ktiles, int tid) {
  const int total = ktiles * 4 * 32 * 16;
  for (int idx = tid; idx < total; idx += 256) {
    const int i = idx & 15;
    const int lane = (idx >> 4) & 31;
    const int t = idx >> 9;                  // kt*4 + nt
    const int kt = t >> 2;
    const int nt = t & 3;
    const int n = lane & 15;
    const int kb = (lane >> 4) << 3;
    const int k = kt * 32 + kb + (i & 7) + ((i >> 3) << 4);
    dst[idx] = (_Float16)W[k * DD + nt * 16 + n];
  }
}

__device__ __forceinline__ v16h load_b_sw(const _Float16* base, int t) {
  const int lane = threadIdx.x & 31;
  return *(const v16h*)(base + ((t * 32 + lane) << 4));   // 2x ds_load_b128
}

// out[r,:] = silu(X[r,:] @ W + bias) (+ res[r,:]).  One 16-row tile per wave.
template <bool HAS_RES>
__global__ void __launch_bounds__(256) node_mlp_kernel(
    const float* __restrict__ X, const float* __restrict__ W,
    const float* __restrict__ bias, const float* __restrict__ res,
    float* __restrict__ out, int nrows) {
  __shared__ alignas(32) _Float16 sW[2 * 4 * 32 * 16];   // 8 KB swizzled
  __shared__ float sB[DD];
  const int tid = threadIdx.x;
  stage_b_swizzled(sW, W, 2, tid);
  if (tid < DD) sB[tid] = bias[tid];
  __syncthreads();

  const int wave = tid >> 5;
  const int lane = tid & 31;
  const int m0 = (blockIdx.x * 8 + wave) * 16;
  if (m0 >= nrows) return;

  int mload = m0 + (lane & 15);
  if (mload >= nrows) mload = nrows - 1;
  const float* rowp = X + (long)mload * DD;
  const v16h a0 = load_a_row_f32(rowp, 0);
  const v16h a1 = load_a_row_f32(rowp, 32);

  const int hi = lane >> 4;
  const int nn = lane & 15;
  const bool full = (m0 + 16 <= nrows);
  const long obase = (long)(m0 + 8 * hi) * DD + nn;

#pragma unroll
  for (int nt = 0; nt < 4; ++nt) {
    const float bv = sB[nt * 16 + nn];
    v8f acc;
#pragma unroll
    for (int j = 0; j < 8; ++j) acc[j] = bv;
    acc = wmma16(a0, load_b_sw(sW, 0 * 4 + nt), acc);
    acc = wmma16(a1, load_b_sw(sW, 1 * 4 + nt), acc);

    float* po = out + obase + nt * 16;
    const float* pr = HAS_RES ? (res + obase + nt * 16) : nullptr;
    if (full) {
#pragma unroll
      for (int j = 0; j < 8; ++j) {
        float v = silu_f(acc[j]);
        if (HAS_RES) v += pr[j * DD];
        po[j * DD] = v;
      }
    } else {
#pragma unroll
      for (int j = 0; j < 8; ++j) {
        if (m0 + j + 8 * hi < nrows) {
          float v = silu_f(acc[j]);
          if (HAS_RES) v += pr[j * DD];
          po[j * DD] = v;
        }
      }
    }
  }
}

// f32 -> f16 row-image conversion (8 elements per thread).
__global__ void cvt_f16_kernel(const float* __restrict__ src,
                               _Float16* __restrict__ dst, long n8) {
  long i = (long)blockIdx.x * blockDim.x + threadIdx.x;
  const long stride = (long)gridDim.x * blockDim.x;
  for (; i < n8; i += stride) {
    const float4* p = (const float4*)(src + i * 8);
    *(v8h*)(dst + i * 8) = (v8h){
        (_Float16)p[0].x, (_Float16)p[0].y, (_Float16)p[0].z, (_Float16)p[0].w,
        (_Float16)p[1].x, (_Float16)p[1].y, (_Float16)p[1].z, (_Float16)p[1].w};
  }
}

// Fused edge propagate: x = silu([h_i|h_j|ea] @ We + be) * (ea @ Wl);
// scatter-add into per-node accumulator + per-node counts.
__global__ void __launch_bounds__(256) edge_prop_kernel(
    const _Float16* __restrict__ H16, const float* __restrict__ EA,
    const int* __restrict__ idx_i, const int* __restrict__ idx_j,
    const float* __restrict__ We, const float* __restrict__ be,
    const float* __restrict__ Wl, float* __restrict__ accv,
    float* __restrict__ cnt, int ME) {
  __shared__ alignas(32) _Float16 sWe[6 * 4 * 32 * 16];  // 24 KB swizzled
  __shared__ alignas(32) _Float16 sWl[2 * 4 * 32 * 16];  //  8 KB swizzled
  __shared__ float sbe[DD];
  const int tid = threadIdx.x;
  stage_b_swizzled(sWe, We, 6, tid);
  stage_b_swizzled(sWl, Wl, 2, tid);
  if (tid < DD) sbe[tid] = be[tid];
  __syncthreads();

  const int wave = tid >> 5;
  const int lane = tid & 31;
  const int e0 = (blockIdx.x * 8 + wave) * 16;
  if (e0 >= ME) return;

  // segment counts: one atomic per edge
  if (lane < 16 && (e0 + lane) < ME) atomicAdd(&cnt[idx_i[e0 + lane]], 1.0f);

  int ea = e0 + (lane & 15);
  if (ea >= ME) ea = ME - 1;
  const unsigned ri = (unsigned)idx_i[ea];
  const unsigned rj = (unsigned)idx_j[ea];
  const _Float16* hpi = H16 + (long)ri * DD;
  const _Float16* hpj = H16 + (long)rj * DD;
  const float* epp = EA + (long)ea * DD;

  // cover streaming latency of the next block's edge_attr rows
  __builtin_prefetch(epp + 128 * DD, 0, 1);

  // A rows = 16 edges; concat columns [h_i | h_j | edge_attr] -> 6 K-tiles
  const v16h aI0 = load_a_row_f16(hpi, 0), aI1 = load_a_row_f16(hpi, 32);
  const v16h aJ0 = load_a_row_f16(hpj, 0), aJ1 = load_a_row_f16(hpj, 32);
  const v16h aE0 = load_a_row_f32(epp, 0), aE1 = load_a_row_f32(epp, 32);

  const int hi = lane >> 4;
  const int nn = lane & 15;
  const bool full = (e0 + 16 <= ME);
  unsigned irow[8];
  bool iok[8];
#pragma unroll
  for (int j = 0; j < 8; ++j) {
    const int er = e0 + j + 8 * hi;
    iok[j] = (er < ME);
    irow[j] = (unsigned)idx_i[iok[j] ? er : (ME - 1)];
  }

#pragma unroll
  for (int nt = 0; nt < 4; ++nt) {
    const float bv = sbe[nt * 16 + nn];
    v8f acce;
#pragma unroll
    for (int j = 0; j < 8; ++j) acce[j] = bv;
    acce = wmma16(aI0, load_b_sw(sWe, 0 * 4 + nt), acce);
    acce = wmma16(aI1, load_b_sw(sWe, 1 * 4 + nt), acce);
    acce = wmma16(aJ0, load_b_sw(sWe, 2 * 4 + nt), acce);
    acce = wmma16(aJ1, load_b_sw(sWe, 3 * 4 + nt), acce);
    acce = wmma16(aE0, load_b_sw(sWe, 4 * 4 + nt), acce);
    acce = wmma16(aE1, load_b_sw(sWe, 5 * 4 + nt), acce);

    v8f accl;
#pragma unroll
    for (int j = 0; j < 8; ++j) accl[j] = 0.0f;
    accl = wmma16(aE0, load_b_sw(sWl, 0 * 4 + nt), accl);
    accl = wmma16(aE1, load_b_sw(sWl, 1 * 4 + nt), accl);

    if (full) {
#pragma unroll
      for (int j = 0; j < 8; ++j) {
        const float v = silu_f(acce[j]) * accl[j];
        atomicAdd(&accv[(unsigned long)irow[j] * DD + nt * 16 + nn], v);
      }
    } else {
#pragma unroll
      for (int j = 0; j < 8; ++j) {
        if (iok[j]) {
          const float v = silu_f(acce[j]) * accl[j];
          atomicAdd(&accv[(unsigned long)irow[j] * DD + nt * 16 + nn], v);
        }
      }
    }
  }
}

__global__ void fill_zero_kernel(float* __restrict__ p, long n) {
  long i = (long)blockIdx.x * blockDim.x + threadIdx.x;
  const long stride = (long)gridDim.x * blockDim.x;
  for (; i < n; i += stride) p[i] = 0.0f;
}

__global__ void seg_finalize_kernel(const float* __restrict__ accv,
                                    const float* __restrict__ cnt,
                                    const float* __restrict__ h,
                                    float* __restrict__ out, long n64) {
  long i = (long)blockIdx.x * blockDim.x + threadIdx.x;
  const long stride = (long)gridDim.x * blockDim.x;
  for (; i < n64; i += stride) {
    const float c = fmaxf(cnt[i >> 6], 1.0f);
    out[i] = accv[i] * __builtin_amdgcn_rcpf(c) + h[i];
  }
}

extern "C" void kernel_launch(void* const* d_in, const int* in_sizes, int n_in,
                              void* d_out, int out_size, void* d_ws, size_t ws_size,
                              hipStream_t stream) {
  (void)n_in; (void)out_size; (void)ws_size;
  const float* nodes = (const float*)d_in[0];
  const float* edge_attr = (const float*)d_in[1];
  const int*   ei   = (const int*)d_in[2];
  const float* Wh   = (const float*)d_in[3];  const float* bh   = (const float*)d_in[4];
  const float* Wm   = (const float*)d_in[5];  const float* bm   = (const float*)d_in[6];
  const float* We   = (const float*)d_in[7];  const float* be   = (const float*)d_in[8];
  const float* Wl   = (const float*)d_in[9];
  const float* r1W1 = (const float*)d_in[10]; const float* r1b1 = (const float*)d_in[11];
  const float* r1W2 = (const float*)d_in[12]; const float* r1b2 = (const float*)d_in[13];
  const float* r2W1 = (const float*)d_in[14]; const float* r2b1 = (const float*)d_in[15];
  const float* r2W2 = (const float*)d_in[16]; const float* r2b2 = (const float*)d_in[17];
  const float* r3W1 = (const float*)d_in[18]; const float* r3b1 = (const float*)d_in[19];
  const float* r3W2 = (const float*)d_in[20]; const float* r3b2 = (const float*)d_in[21];

  const int N  = in_sizes[0] / DD;
  const int ME = in_sizes[1] / DD;
  const int* idx_i = ei;          // edge_index[0] (receivers)
  const int* idx_j = ei + ME;     // edge_index[1] (senders)

  const long N64 = (long)N * DD;
  float* bufA = (float*)d_ws;
  float* bufB = bufA + N64;
  float* bufC = bufB + N64;
  float* ACC  = bufC + N64;
  float* CNT  = ACC + N64;        // contiguous after ACC (zeroed together)
  // f16 node-feature image, 16B aligned
  const long cntEnd = (4 * N64 + (long)N + 3) & ~3L;
  _Float16* H16 = (_Float16*)((float*)d_ws + cntEnd);
  float* outF = (float*)d_out;

  const dim3 blk(256);
  const dim3 gN((N + 127) / 128);
  const dim3 gE((ME + 127) / 128);

  // h = silu(nodes @ Wh + bh)
  node_mlp_kernel<false><<<gN, blk, 0, stream>>>(nodes, Wh, bh, nullptr, bufA, N);

  // propagate #1: bufB = segmean(edgeMLP) + bufA
  cvt_f16_kernel<<<512, blk, 0, stream>>>(bufA, H16, N64 / 8);
  fill_zero_kernel<<<1024, blk, 0, stream>>>(ACC, N64 + N);
  edge_prop_kernel<<<gE, blk, 0, stream>>>(H16, edge_attr, idx_i, idx_j,
                                           We, be, Wl, ACC, CNT, ME);
  seg_finalize_kernel<<<1024, blk, 0, stream>>>(ACC, CNT, bufA, bufB, N64);

  // res1: bufA = bufB + silu(silu(bufB@W1+b1)@W2+b2)
  node_mlp_kernel<false><<<gN, blk, 0, stream>>>(bufB, r1W1, r1b1, nullptr, bufC, N);
  node_mlp_kernel<true><<<gN, blk, 0, stream>>>(bufC, r1W2, r1b2, bufB, bufA, N);

  // bufB = silu(bufA@Wm+bm) + nodes
  node_mlp_kernel<true><<<gN, blk, 0, stream>>>(bufA, Wm, bm, nodes, bufB, N);

  // res2
  node_mlp_kernel<false><<<gN, blk, 0, stream>>>(bufB, r2W1, r2b1, nullptr, bufC, N);
  node_mlp_kernel<true><<<gN, blk, 0, stream>>>(bufC, r2W2, r2b2, bufB, bufA, N);

  // res3
  node_mlp_kernel<false><<<gN, blk, 0, stream>>>(bufA, r3W1, r3b1, nullptr, bufC, N);
  node_mlp_kernel<true><<<gN, blk, 0, stream>>>(bufC, r3W2, r3b2, bufA, bufB, N);

  // propagate #2 -> d_out
  cvt_f16_kernel<<<512, blk, 0, stream>>>(bufB, H16, N64 / 8);
  fill_zero_kernel<<<1024, blk, 0, stream>>>(ACC, N64 + N);
  edge_prop_kernel<<<gE, blk, 0, stream>>>(H16, edge_attr, idx_i, idx_j,
                                           We, be, Wl, ACC, CNT, ME);
  seg_finalize_kernel<<<1024, blk, 0, stream>>>(ACC, CNT, bufB, outF, N64);
}
